// Hourglass_nonhistory_16904991277606
// MI455X (gfx1250) — compile-verified, matrix operand order checked by reading, not run on hardware
//
#include <hip/hip_runtime.h>
#include <hip/hip_bf16.h>

// ---------------------------------------------------------------------------
// Hourglass network on MI455X (gfx1250, wave32).
// All GEMM-like work (1x1 convs, implicit-GEMM 3x3 convs, attention
// projections, logits, attn*V) runs through one WMMA kernel using
// v_wmma_f32_16x16x32_bf16 (fp32 data converted to bf16 on load, f32 acc).
// Block tile 64Mx128Nx32K, 8 waves, each wave does 4 WMMAs per K-step.
// Software-pipelined: next K-tile is prefetched into registers while the
// current tile's WMMAs execute; masking happens in registers so LDS stores
// are always vectorized (ds_store_b128 / packed ds_store_b32).
// Elementwise kernels: maxpool2, softmax over L=24, upsample2+add.
// Workspace arena use: ~245 MB.
// ---------------------------------------------------------------------------

typedef __bf16 bf16;
typedef __attribute__((ext_vector_type(8)))  __bf16  bf16x8;
typedef __attribute__((ext_vector_type(16))) __bf16  bf16x16;
typedef __attribute__((ext_vector_type(8)))  float   floatx8;
typedef __attribute__((ext_vector_type(4)))  float   floatx4;
typedef __attribute__((ext_vector_type(4)))  unsigned uintx4;

__device__ __forceinline__ unsigned bfround(unsigned u) {
    return u + 0x7FFFu + ((u >> 16) & 1u);   // round-to-nearest-even
}
__device__ __forceinline__ bf16 f2bf(float x) {
    unsigned short h = (unsigned short)(bfround(__builtin_bit_cast(unsigned, x)) >> 16);
    return __builtin_bit_cast(bf16, h);
}
// pack two floats as {bf16(lo), bf16(hi)} in one dword (little-endian order)
__device__ __forceinline__ unsigned pack2(float lo, float hi) {
    unsigned a = bfround(__builtin_bit_cast(unsigned, lo)) >> 16;
    unsigned b = bfround(__builtin_bit_cast(unsigned, hi)) & 0xFFFF0000u;
    return a | b;
}

#define BM 64
#define BN 128
#define BKT 32
#define LDSS 40   // padded LDS row stride (bf16); rows 80B, 16B chunks stay aligned

__device__ __forceinline__ bf16x16 load_frag(const bf16* row, int khalf) {
    // WMMA 16-bit A/B operand layout (ISA 7.12.2): lanes 0-15 hold K 0..7
    // in v0-3 and K 16..23 in v4-7; lanes 16-31 hold K 8..15 and 24..31.
    bf16x8 lo = *(const bf16x8*)(row + (khalf << 3));
    bf16x8 hi = *(const bf16x8*)(row + 16 + (khalf << 3));
    bf16x16 r;
#pragma unroll
    for (int i = 0; i < 8; ++i) { r[i] = lo[i]; r[i + 8] = hi[i]; }
    return r;
}

// register staging for one 64x32 A tile + 128x32 B tile (per thread slice)
struct TileRegs {
    floatx4 a0, a1;           // A: 8 contiguous k for one m row
    floatx4 b0, b1, b2, b3;   // B: two halves of the 128x32 tile
};

// ---- A: [M,K] row-major, 8 contiguous k per thread -------------------------
__device__ __forceinline__ void load_a_regs(TileRegs& r, const float* Ab, int m0,
                                            int M, int K, int lda, int kt, int tid,
                                            bool fast /* fullM && kfull */)
{
    const int mm = tid >> 2;
    const int kk = (tid & 3) << 3;
    const int gm = m0 + mm;
    if (fast) {
        const float* src = Ab + (long)gm * lda + kt + kk;
        r.a0 = *(const floatx4*)(src);
        r.a1 = *(const floatx4*)(src + 4);
    } else {
        const int  gmc = gm < M ? gm : (M - 1);
        const bool okm = gm < M;
        const float* src = Ab + (long)gmc * lda;
#pragma unroll
        for (int j = 0; j < 8; ++j) {
            int gk  = kt + kk + j;
            int gkc = gk < K ? gk : (K - 1);
            float t = src[gkc];                     // clamped, unconditional
            float v = (okm && gk < K) ? t : 0.f;
            if (j < 4) r.a0[j] = v; else r.a1[j - 4] = v;
        }
    }
}
__device__ __forceinline__ void store_a_lds(const TileRegs& r, bf16* lsa, int tid)
{
    const int mm = tid >> 2;
    const int kk = (tid & 3) << 3;
    uintx4 p;
    p[0] = pack2(r.a0[0], r.a0[1]); p[1] = pack2(r.a0[2], r.a0[3]);
    p[2] = pack2(r.a1[0], r.a1[1]); p[3] = pack2(r.a1[2], r.a1[3]);
    *(uintx4*)(lsa + mm * LDSS + kk) = p;
}

// ---- B bmode1: B^T[n][k] row-major (k contiguous) --------------------------
__device__ __forceinline__ void load_b1_regs(TileRegs& r, const float* Bb, int n0,
                                             int N, int K, int ldb, int kt, int tid,
                                             bool fast /* fullN && kfull */)
{
#pragma unroll
    for (int h = 0; h < 2; ++h) {
        int t2 = h * 256 + tid;
        int nn = t2 >> 2;              // 0..127
        int kk = (t2 & 3) << 3;
        int gn = n0 + nn;
        floatx4 f0, f1;
        if (fast) {
            const float* src = Bb + (long)gn * ldb + kt + kk;
            f0 = *(const floatx4*)(src);
            f1 = *(const floatx4*)(src + 4);
        } else {
            const int  gnc = gn < N ? gn : (N - 1);
            const bool okn = gn < N;
            const float* src = Bb + (long)gnc * ldb;
#pragma unroll
            for (int j = 0; j < 8; ++j) {
                int gk  = kt + kk + j;
                int gkc = gk < K ? gk : (K - 1);
                float t = src[gkc];
                float v = (okn && gk < K) ? t : 0.f;
                if (j < 4) f0[j] = v; else f1[j - 4] = v;
            }
        }
        if (h == 0) { r.b0 = f0; r.b1 = f1; } else { r.b2 = f0; r.b3 = f1; }
    }
}
__device__ __forceinline__ void store_b1_lds(const TileRegs& r, bf16* lsb, int tid)
{
#pragma unroll
    for (int h = 0; h < 2; ++h) {
        int t2 = h * 256 + tid;
        int nn = t2 >> 2;
        int kk = (t2 & 3) << 3;
        floatx4 f0 = h ? r.b2 : r.b0;
        floatx4 f1 = h ? r.b3 : r.b1;
        uintx4 p;
        p[0] = pack2(f0[0], f0[1]); p[1] = pack2(f0[2], f0[3]);
        p[2] = pack2(f1[0], f1[1]); p[3] = pack2(f1[2], f1[3]);
        *(uintx4*)(lsb + nn * LDSS + kk) = p;
    }
}

// ---- B bmode0: B[k][n], n contiguous (NCHW activations) --------------------
__device__ __forceinline__ void load_b0_regs(TileRegs& r, const float* Bb, int n0,
                                             int N, int K, int ldb, int kt, int tid,
                                             bool fast /* fullN && kfull */)
{
#pragma unroll
    for (int h = 0; h < 2; ++h) {
        int t2 = h * 256 + tid;
        int ng = t2 & 31;
        int kp = t2 >> 5;              // 0..15
        int kk = kp << 1;
        int nb = ng << 2;
        floatx4 f0, f1;
        if (fast) {
            const float* r0 = Bb + (long)(kt + kk) * ldb + n0 + nb;
            f0 = *(const floatx4*)(r0);
            f1 = *(const floatx4*)(r0 + ldb);
        } else {
            int gk0 = kt + kk, gk1 = gk0 + 1;
            int gk0c = gk0 < K ? gk0 : K - 1;
            int gk1c = gk1 < K ? gk1 : K - 1;
            const float* r0 = Bb + (long)gk0c * ldb;
            const float* r1 = Bb + (long)gk1c * ldb;
#pragma unroll
            for (int j = 0; j < 4; ++j) {
                int gn  = n0 + nb + j;
                int gnc = gn < N ? gn : N - 1;
                float t0 = r0[gnc];
                float t1 = r1[gnc];
                f0[j] = (gk0 < K && gn < N) ? t0 : 0.f;
                f1[j] = (gk1 < K && gn < N) ? t1 : 0.f;
            }
        }
        if (h == 0) { r.b0 = f0; r.b1 = f1; } else { r.b2 = f0; r.b3 = f1; }
    }
}

// ---- B bmode2: implicit 3x3 conv gather ------------------------------------
__device__ __forceinline__ void load_b2_regs(TileRegs& r, const float* Bb, int n0,
                                             int N, int K, int H, int W, int kt, int tid)
{
#pragma unroll
    for (int h = 0; h < 2; ++h) {
        int t2 = h * 256 + tid;
        int ng = t2 & 31;
        int kp = t2 >> 5;
        int kk = kp << 1;
        int nb = ng << 2;
        int gk0 = kt + kk, gk1 = gk0 + 1;
        int gk0c = gk0 < K ? gk0 : K - 1;
        int gk1c = gk1 < K ? gk1 : K - 1;
        int ci0 = gk0c / 9, rr0 = gk0c - ci0 * 9;
        int dy0 = rr0 / 3 - 1, dx0 = rr0 - (rr0 / 3) * 3 - 1;
        int ci1 = gk1c / 9, rr1 = gk1c - ci1 * 9;
        int dy1 = rr1 / 3 - 1, dx1 = rr1 - (rr1 / 3) * 3 - 1;
        floatx4 f0, f1;
#pragma unroll
        for (int j = 0; j < 4; ++j) {
            int gn  = n0 + nb + j;
            int gnc = gn < N ? gn : N - 1;
            int hh  = gnc / W, ww = gnc - hh * W;
            {
                int hy = hh + dy0, wx = ww + dx0;
                bool ok = (gk0 < K) && (gn < N) &&
                          ((unsigned)hy < (unsigned)H) && ((unsigned)wx < (unsigned)W);
                int hyc = hy < 0 ? 0 : (hy >= H ? H - 1 : hy);
                int wxc = wx < 0 ? 0 : (wx >= W ? W - 1 : wx);
                float t = Bb[((long)ci0 * H + hyc) * W + wxc];
                f0[j] = ok ? t : 0.f;
            }
            {
                int hy = hh + dy1, wx = ww + dx1;
                bool ok = (gk1 < K) && (gn < N) &&
                          ((unsigned)hy < (unsigned)H) && ((unsigned)wx < (unsigned)W);
                int hyc = hy < 0 ? 0 : (hy >= H ? H - 1 : hy);
                int wxc = wx < 0 ? 0 : (wx >= W ? W - 1 : wx);
                float t = Bb[((long)ci1 * H + hyc) * W + wxc];
                f1[j] = ok ? t : 0.f;
            }
        }
        if (h == 0) { r.b0 = f0; r.b1 = f1; } else { r.b2 = f0; r.b3 = f1; }
    }
}

// shared store for bmode0/2 (transposed): pack the two k rows per n
__device__ __forceinline__ void store_b0_lds(const TileRegs& r, bf16* lsb, int tid)
{
#pragma unroll
    for (int h = 0; h < 2; ++h) {
        int t2 = h * 256 + tid;
        int ng = t2 & 31;
        int kp = t2 >> 5;
        int kk = kp << 1;
        int nb = ng << 2;
        floatx4 f0 = h ? r.b2 : r.b0;
        floatx4 f1 = h ? r.b3 : r.b1;
#pragma unroll
        for (int j = 0; j < 4; ++j)
            *(unsigned*)(lsb + (nb + j) * LDSS + kk) = pack2(f0[j], f1[j]);
    }
}

// D[m][n] = relu?( alpha * sum_k A[m][k]*B[k][n] + bias + add )
// bmode: 0 = B[k][n], n contiguous; 1 = B^T[n][k], k contiguous; 2 = implicit 3x3
// biasmode: 0 none, 1 per-m, 2 per-n
__global__ __launch_bounds__(256)
void wmma_gemm_kernel(const float* __restrict__ A, const float* __restrict__ Bsrc,
                      const float* __restrict__ bias, const float* __restrict__ addsrc,
                      float* __restrict__ out,
                      int M, int N, int K, int lda, int ldb,
                      int bmode, int H, int W,
                      long a_bs, long b_bs, long out_bs, long out_off,
                      long add_bs, long add_off,
                      float alpha, int relu, int biasmode)
{
    __shared__ alignas(16) bf16 lsa[BM * LDSS];
    __shared__ alignas(16) bf16 lsb[BN * LDSS];

    const int tid  = threadIdx.x;
    const int b    = blockIdx.z;
    const int m0   = blockIdx.y * BM;
    const int n0   = blockIdx.x * BN;
    const float* Ab = A    + (long)b * a_bs;
    const float* Bb = Bsrc + (long)b * b_bs;

    const int wave  = tid >> 5;
    const int lane  = tid & 31;
    const int wm    = wave & 3;   // 4 waves along M (16 rows each)
    const int wn    = wave >> 2;  // 2 waves along N (64 cols each)
    const int khalf = lane >> 4;
    const int lrow  = lane & 15;

    const bool fullM = (m0 + BM <= M);
    const bool fullN = (n0 + BN <= N);

    floatx8 acc[4] = {{}, {}, {}, {}};
    TileRegs r;

    auto loadB = [&](int kt, bool kfull) {
        if (bmode == 1)      load_b1_regs(r, Bb, n0, N, K, ldb, kt, tid, fullN && kfull);
        else if (bmode == 0) load_b0_regs(r, Bb, n0, N, K, ldb, kt, tid, fullN && kfull);
        else                 load_b2_regs(r, Bb, n0, N, K, H, W, kt, tid);
    };
    auto storeB = [&]() {
        if (bmode == 1) store_b1_lds(r, lsb, tid);
        else            store_b0_lds(r, lsb, tid);
    };
    auto compute = [&]() {
        bf16x16 af = load_frag(lsa + (wm * 16 + lrow) * LDSS, khalf);
#pragma unroll
        for (int t = 0; t < 4; ++t) {
            bf16x16 bt = load_frag(lsb + (wn * 64 + t * 16 + lrow) * LDSS, khalf);
            acc[t] = __builtin_amdgcn_wmma_f32_16x16x32_bf16(false, af, false, bt,
                                                             (short)0, acc[t], false, false);
        }
    };

    const int Kmain = K & ~(BKT - 1);
    if (Kmain > 0) {
        // prologue: first tile into registers
        load_a_regs(r, Ab, m0, M, K, lda, 0, tid, fullM);
        loadB(0, true);
        for (int kt = 0; kt < Kmain; kt += BKT) {
            store_a_lds(r, lsa, tid);
            storeB();
            __syncthreads();
            int ktn = kt + BKT;
            if (ktn < Kmain) {               // prefetch next tile; latency overlaps WMMAs
                load_a_regs(r, Ab, m0, M, K, lda, ktn, tid, fullM);
                loadB(ktn, true);
            }
            compute();
            __syncthreads();
        }
    }
    if (Kmain < K) {                         // masked K tail (K not multiple of 32)
        load_a_regs(r, Ab, m0, M, K, lda, Kmain, tid, false);
        loadB(Kmain, false);
        store_a_lds(r, lsa, tid);
        storeB();
        __syncthreads();
        compute();
    }

    // ---- epilogue (C/D layout: lanes 0-15 -> M=r, lanes 16-31 -> M=r+8, N=lane&15)
    float* outb = out + (long)b * out_bs + out_off;
    const float* addb = addsrc ? (addsrc + (long)b * add_bs + add_off) : nullptr;
#pragma unroll
    for (int r8 = 0; r8 < 8; ++r8) {
        int gm = m0 + wm * 16 + r8 + (khalf << 3);
        if (gm >= M) continue;
        float bm = (biasmode == 1) ? bias[gm] : 0.f;
#pragma unroll
        for (int t = 0; t < 4; ++t) {
            int gn = n0 + wn * 64 + t * 16 + lrow;
            if (gn < N) {
                float v = acc[t][r8] * alpha + bm;
                if (biasmode == 2) v += bias[gn];
                if (addb) v += addb[(long)gm * N + gn];
                if (relu) v = v > 0.f ? v : 0.f;
                outb[(long)gm * N + gn] = v;
            }
        }
    }
}

// out[b,c,ho,wo] = max 2x2
__global__ void maxpool2_kernel(const float* __restrict__ in, float* __restrict__ out,
                                int C, int Hi, int Wi, long total)
{
    long i = blockIdx.x * (long)blockDim.x + threadIdx.x;
    if (i >= total) return;
    int Ho = Hi >> 1, Wo = Wi >> 1;
    long t = i;
    int wo = (int)(t % Wo); t /= Wo;
    int ho = (int)(t % Ho); t /= Ho;
    int c  = (int)(t % C);
    int b  = (int)(t / C);
    const float* p = in + (((long)b * C + c) * Hi + 2 * ho) * Wi + 2 * wo;
    float m0 = fmaxf(p[0], p[1]);
    float m1 = fmaxf(p[Wi], p[Wi + 1]);
    out[i] = fmaxf(m0, m1);
}

// softmax along L=24 of logits[b][l][hw] (stride HW), in place
__global__ void softmax24_kernel(float* __restrict__ lg, int HW, long bs, long total)
{
    long i = blockIdx.x * (long)blockDim.x + threadIdx.x;
    if (i >= total) return;
    int b = (int)(i / HW);
    int n = (int)(i - (long)b * HW);
    float* p = lg + (long)b * bs + n;
    float mx = -3.402823466e38f;
#pragma unroll
    for (int l = 0; l < 24; ++l) mx = fmaxf(mx, p[(long)l * HW]);
    float e[24], s = 0.f;
#pragma unroll
    for (int l = 0; l < 24; ++l) { e[l] = __expf(p[(long)l * HW] - mx); s += e[l]; }
    float inv = 1.f / s;
#pragma unroll
    for (int l = 0; l < 24; ++l) p[(long)l * HW] = e[l] * inv;
}

// out = up + upsample2(low); out may alias up (element-wise same index)
__global__ void add_upsample_kernel(const float* __restrict__ up, const float* __restrict__ low,
                                    float* __restrict__ out, int C, int Ho, int Wo,
                                    long up_bs, long low_bs, long out_bs, long out_off, long total)
{
    long i = blockIdx.x * (long)blockDim.x + threadIdx.x;
    if (i >= total) return;
    long t = i;
    int x = (int)(t % Wo); t /= Wo;
    int y = (int)(t % Ho); t /= Ho;
    int c = (int)(t % C);
    int b = (int)(t / C);
    int Hl = Ho >> 1, Wl = Wo >> 1;
    float u = up[(long)b * up_bs + ((long)c * Ho + y) * Wo + x];
    float l = low[(long)b * low_bs + ((long)c * Hl + (y >> 1)) * Wl + (x >> 1)];
    out[(long)b * out_bs + out_off + ((long)c * Ho + y) * Wo + x] = u + l;
}

// ---------------------------------------------------------------------------
// Host-side orchestration
// ---------------------------------------------------------------------------
namespace {

struct ResW { const float *c1w,*c1b,*c2w,*c2b,*c3w,*c3b,*skw,*skb; };

inline void gemm(hipStream_t s, const float* A, const float* B, const float* bias,
                 const float* add, float* out, int M, int N, int K, int lda, int ldb,
                 int bmode, int H, int W, long a_bs, long b_bs,
                 long out_bs, long out_off, long add_bs, long add_off,
                 float alpha, int relu, int biasmode)
{
    dim3 g((N + BN - 1) / BN, (M + BM - 1) / BM, 8);
    wmma_gemm_kernel<<<g, 256, 0, s>>>(A, B, bias, add, out, M, N, K, lda, ldb,
                                       bmode, H, W, a_bs, b_bs, out_bs, out_off,
                                       add_bs, add_off, alpha, relu, biasmode);
}

// out[b, 0..255, H, W] (with out_bs/out_off) = residual(w, in)
inline void residual_seq(hipStream_t s, const ResW& w, const float* in, long in_bs,
                         int Cin, int H, int W, float* t1, float* t2, float* sk,
                         float* out, long out_bs, long out_off)
{
    long HW = (long)H * W;
    // c1: 1x1, Cin -> 128, relu
    gemm(s, w.c1w, in, w.c1b, nullptr, t1, 128, (int)HW, Cin, Cin, (int)HW,
         0, H, W, 0, in_bs, 128 * HW, 0, 0, 0, 1.f, 1, 1);
    // c2: 3x3, 128 -> 128, relu (implicit GEMM, K = 128*9)
    gemm(s, w.c2w, t1, w.c2b, nullptr, t2, 128, (int)HW, 128 * 9, 128 * 9, 0,
         2, H, W, 0, 128 * HW, 128 * HW, 0, 0, 0, 1.f, 1, 1);
    // skip path
    const float* addp; long abs_;
    if (w.skw) {
        gemm(s, w.skw, in, w.skb, nullptr, sk, 256, (int)HW, Cin, Cin, (int)HW,
             0, H, W, 0, in_bs, 256 * HW, 0, 0, 0, 1.f, 0, 1);
        addp = sk; abs_ = 256 * HW;
    } else {
        addp = in; abs_ = in_bs;   // identity (Cin == 256)
    }
    // c3: 1x1, 128 -> 256, + skip
    gemm(s, w.c3w, t2, w.c3b, addp, out, 256, (int)HW, 128, 128, (int)HW,
         0, H, W, 0, 128 * HW, out_bs, out_off, abs_, 0, 1.f, 0, 1);
}

// attention at one level; low2 lives in fuse[ch 0..255]; writes attn_val into fuse[ch 256..511]
inline void attention_seq(hipStream_t s, const float* we,
                          const float* Wq, const float* bq,
                          const float* Wk, const float* bk,
                          const float* Wv, const float* bv,
                          float* fuse, int H, int W,
                          float* krow, float* vT, float* qT, float* logits)
{
    long HW = (long)H * W;
    // k (row-major [B,24,256]): A = word_embed[b] (24x768), B^T = Wk (256x768), bias per-n
    gemm(s, we, Wk, bk, nullptr, krow, 24, 256, 768, 768, 768,
         1, 0, 0, 24L * 768, 0, 24L * 256, 0, 0, 0, 1.f, 0, 2);
    // v^T ([B,256,24]): A = Wv (256x768), B^T = word_embed[b], bias per-m
    gemm(s, Wv, we, bv, nullptr, vT, 256, 24, 768, 768, 768,
         1, 0, 0, 0, 24L * 768, 256L * 24, 0, 0, 0, 1.f, 0, 1);
    // q^T ([B,256,HW]): 1x1-conv form over low2 (fuse ch0-255, batch stride 512*HW)
    gemm(s, Wq, fuse, bq, nullptr, qT, 256, (int)HW, 256, 256, (int)HW,
         0, H, W, 0, 512 * HW, 256 * HW, 0, 0, 0, 1.f, 0, 1);
    // logits^T ([B,24,HW]) = k[b] (24x256) x q^T[b] (256xHW), scaled by 1/sqrt(256)
    gemm(s, krow, qT, nullptr, nullptr, logits, 24, (int)HW, 256, 256, (int)HW,
         0, H, W, 24L * 256, 256 * HW, 24 * HW, 0, 0, 0, 1.f / 16.f, 0, 0);
    // softmax over L=24
    long tot = 8L * HW;
    softmax24_kernel<<<(int)((tot + 255) / 256), 256, 0, s>>>(logits, (int)HW, 24 * HW, tot);
    // attn_val^T ([B,256,HW]) = v^T[b] (256x24) x attn^T[b] (24xHW) -> fuse ch256-511
    gemm(s, vT, logits, nullptr, nullptr, fuse, 256, (int)HW, 24, 24, (int)HW,
         0, H, W, 256L * 24, 24 * HW, 512 * HW, 256 * HW, 0, 0, 1.f, 0, 0);
}

} // namespace

extern "C" void kernel_launch(void* const* d_in, const int* in_sizes, int n_in,
                              void* d_out, int out_size, void* d_ws, size_t ws_size,
                              hipStream_t stream)
{
    (void)in_sizes; (void)n_in; (void)out_size; (void)ws_size;
    auto P = [&](int i) { return (const float*)d_in[i]; };
    const float* x  = P(0);   // [8,256,96,96]
    const float* we = P(1);   // [8,24,768]

    // params flattened in setup_inputs() insertion order
    ResW l1_up1  {P(2),P(3),P(4),P(5),P(6),P(7), nullptr,nullptr};
    ResW l1_low1 {P(8),P(9),P(10),P(11),P(12),P(13), nullptr,nullptr};
    ResW l1_low3 {P(14),P(15),P(16),P(17),P(18),P(19),P(20),P(21)};
    ResW l1_low2 {P(22),P(23),P(24),P(25),P(26),P(27), nullptr,nullptr};
    ResW l2_up1  {P(28),P(29),P(30),P(31),P(32),P(33), nullptr,nullptr};
    ResW l2_low1 {P(34),P(35),P(36),P(37),P(38),P(39), nullptr,nullptr};
    ResW l2_low3 {P(40),P(41),P(42),P(43),P(44),P(45),P(46),P(47)};
    ResW l3_up1  {P(48),P(49),P(50),P(51),P(52),P(53), nullptr,nullptr};
    ResW l3_low1 {P(54),P(55),P(56),P(57),P(58),P(59), nullptr,nullptr};
    ResW l3_low3 {P(60),P(61),P(62),P(63),P(64),P(65),P(66),P(67)};
    const float* Wq_w[3] = {P(68), P(70), P(72)};
    const float* Wq_b[3] = {P(69), P(71), P(73)};
    const float* Wk_w[3] = {P(74), P(76), P(78)};
    const float* Wk_b[3] = {P(75), P(77), P(79)};
    const float* Wv_w[3] = {P(80), P(82), P(84)};
    const float* Wv_b[3] = {P(81), P(83), P(85)};

    const long B = 8;
    const long HW96 = 96 * 96, HW48 = 48 * 48, HW24 = 24 * 24, HW12 = 12 * 12;

    // ---- workspace arena (floats) --------------------------------------
    float* wsf = (float*)d_ws;
    long off = 0;
    auto alloc = [&](long n) { float* p = wsf + off; off += n; return p; };
    float* T1     = alloc(B * 128 * HW96);
    float* T2     = alloc(B * 128 * HW96);
    float* SKIPT  = alloc(B * 256 * HW48);
    float* UP1_2  = alloc(B * 256 * HW48);
    float* UP1_1  = alloc(B * 256 * HW24);
    float* PX3    = alloc(B * 256 * HW48);
    float* LOW1_3 = alloc(B * 256 * HW48);
    float* PX2    = alloc(B * 256 * HW24);
    float* LOW1_2 = alloc(B * 256 * HW24);
    float* PX1    = alloc(B * 256 * HW12);
    float* LOW1_1 = alloc(B * 256 * HW12);
    float* FUSE1  = alloc(B * 512 * HW12);
    float* FUSE2  = alloc(B * 512 * HW24);
    float* FUSE3  = alloc(B * 512 * HW48);
    float* KROW   = alloc(B * 24 * 256);
    float* VT     = alloc(B * 256 * 24);
    float* QT     = alloc(B * 256 * HW48);
    float* LOGITS = alloc(B * 24 * HW48);
    float* LOW3   = alloc(B * 256 * HW48);
    // total ~64.2M floats (~245 MiB)

    float* out = (float*)d_out;
    hipStream_t s = stream;

    // ---- level 3 down path ----
    {
        long tot = B * 256 * HW48;
        maxpool2_kernel<<<(int)((tot + 255) / 256), 256, 0, s>>>(x, PX3, 256, 96, 96, tot);
    }
    residual_seq(s, l3_low1, PX3, 256 * HW48, 256, 48, 48, T1, T2, SKIPT,
                 LOW1_3, 256 * HW48, 0);

    // ---- level 2 down path ----
    {
        long tot = B * 256 * HW24;
        maxpool2_kernel<<<(int)((tot + 255) / 256), 256, 0, s>>>(LOW1_3, PX2, 256, 48, 48, tot);
    }
    residual_seq(s, l2_up1, LOW1_3, 256 * HW48, 256, 48, 48, T1, T2, SKIPT,
                 UP1_2, 256 * HW48, 0);
    residual_seq(s, l2_low1, PX2, 256 * HW24, 256, 24, 24, T1, T2, SKIPT,
                 LOW1_2, 256 * HW24, 0);

    // ---- level 1 ----
    {
        long tot = B * 256 * HW12;
        maxpool2_kernel<<<(int)((tot + 255) / 256), 256, 0, s>>>(LOW1_2, PX1, 256, 24, 24, tot);
    }
    residual_seq(s, l1_up1, LOW1_2, 256 * HW24, 256, 24, 24, T1, T2, SKIPT,
                 UP1_1, 256 * HW24, 0);
    residual_seq(s, l1_low1, PX1, 256 * HW12, 256, 12, 12, T1, T2, SKIPT,
                 LOW1_1, 256 * HW12, 0);
    // low2 (lvl1) -> FUSE1 channels 0..255
    residual_seq(s, l1_low2, LOW1_1, 256 * HW12, 256, 12, 12, T1, T2, SKIPT,
                 FUSE1, 512 * HW12, 0);
    // attention lvl1 -> FUSE1 channels 256..511
    attention_seq(s, we, Wq_w[0], Wq_b[0], Wk_w[0], Wk_b[0], Wv_w[0], Wv_b[0],
                  FUSE1, 12, 12, KROW, VT, QT, LOGITS);
    // low3 (lvl1): Cin = 512 (with skip conv)
    residual_seq(s, l1_low3, FUSE1, 512 * HW12, 512, 12, 12, T1, T2, SKIPT,
                 LOW3, 256 * HW12, 0);
    // hourglass(1) output = up1_1 + upsample(low3) -> FUSE2 channels 0..255 (= low2 of lvl2)
    {
        long tot = B * 256 * HW24;
        add_upsample_kernel<<<(int)((tot + 255) / 256), 256, 0, s>>>(
            UP1_1, LOW3, FUSE2, 256, 24, 24,
            256 * HW24, 256 * HW12, 512 * HW24, 0, tot);
    }

    // ---- level 2 up path ----
    attention_seq(s, we, Wq_w[1], Wq_b[1], Wk_w[1], Wk_b[1], Wv_w[1], Wv_b[1],
                  FUSE2, 24, 24, KROW, VT, QT, LOGITS);
    residual_seq(s, l2_low3, FUSE2, 512 * HW24, 512, 24, 24, T1, T2, SKIPT,
                 LOW3, 256 * HW24, 0);
    // hourglass(2) output -> FUSE3 channels 0..255 (= low2 of lvl3)
    {
        long tot = B * 256 * HW48;
        add_upsample_kernel<<<(int)((tot + 255) / 256), 256, 0, s>>>(
            UP1_2, LOW3, FUSE3, 256, 48, 48,
            256 * HW48, 256 * HW24, 512 * HW48, 0, tot);
    }

    // ---- level 3 up path ----
    attention_seq(s, we, Wq_w[2], Wq_b[2], Wk_w[2], Wk_b[2], Wv_w[2], Wv_b[2],
                  FUSE3, 48, 48, KROW, VT, QT, LOGITS);
    residual_seq(s, l3_low3, FUSE3, 512 * HW48, 512, 48, 48, T1, T2, SKIPT,
                 LOW3, 256 * HW48, 0);

    // up1_3 computed last, straight into d_out (avoids a 75 MB buffer)
    residual_seq(s, l3_up1, x, 256 * HW96, 256, 96, 96, T1, T2, SKIPT,
                 out, 256 * HW96, 0);
    // final: out = up1_3 + upsample(low3_3)  (in-place accumulate, same-index r/w)
    {
        long tot = B * 256 * HW96;
        add_upsample_kernel<<<(int)((tot + 255) / 256), 256, 0, s>>>(
            out, LOW3, out, 256, 96, 96,
            256 * HW96, 256 * HW48, 256 * HW96, 0, tot);
    }
}
